// SelfAttentionBlock_72009421684773
// MI455X (gfx1250) — matrix operand order repaired and checked by reading, not compiled, not run on hardware
//
#include <hip/hip_runtime.h>

#define B_ 4
#define C_ 256
#define N_ 4096

typedef __attribute__((ext_vector_type(16))) __bf16 bf16x16;
typedef __attribute__((ext_vector_type(8)))  float  floatx8;
typedef __attribute__((ext_vector_type(4)))  unsigned int uint32x4;
typedef __attribute__((ext_vector_type(8)))  int  int32x8;
typedef __attribute__((ext_vector_type(4)))  int  int32x4;

#if defined(__has_builtin)
#if __has_builtin(__builtin_amdgcn_tensor_load_to_lds) && \
    __has_builtin(__builtin_amdgcn_s_wait_tensorcnt)
#define USE_TDM 1
#endif
#endif

union FragBF {
  bf16x16 v;
  unsigned short u[16];
  unsigned int   w[8];
};

__device__ __forceinline__ unsigned short f32_to_bf16(float f) {
  unsigned int x = __float_as_uint(f);
  unsigned int r = x + 0x7FFFu + ((x >> 16) & 1u);   // round-to-nearest-even
  return (unsigned short)(r >> 16);
}

__device__ __forceinline__ floatx8 zero8() {
  floatx8 z = {0.f, 0.f, 0.f, 0.f, 0.f, 0.f, 0.f, 0.f};
  return z;
}

// 16-bit A-matrix 16x32 layout (ISA 7.12.2): lane = (m | half<<4),
// VGPR i holds K = base, base+1 with base = (i<4 ? 2i : 16+2(i-4)) + 8*half
__device__ __forceinline__ int a_kbase(int i, int half) {
  return (i < 4 ? 2 * i : 16 + 2 * (i - 4)) + 8 * half;
}
// 16-bit B-matrix 32x16 layout: lane = (n | half<<4), VGPR i holds K = base, base+1 with base = (i<4 ? 2i : 16+2(i-4)) + 8*half
__device__ __forceinline__ int b_kbase(int i, int half) {
  return (i < 4 ? 2 * i : 16 + 2 * (i - 4)) + 8 * half;
}

#ifdef USE_TDM
// 2D tile DMA: global (row-major, 32 rows x 256 bf16, row stride C_) -> LDS.
// D# per ISA 8.3-8.5: group0 = {count, lds_addr, global_addr, type=2},
// group1 = {data_size=2B, tensor_dim0=256, tensor_dim1=4096, tile_dim0=256,
//           tile_dim1=32, tensor_dim0_stride=256}.
__device__ __forceinline__ void tdm_load_tile32x256(unsigned lds_addr,
                                                    const void* gptr) {
  unsigned long long ga = (unsigned long long)(size_t)gptr;
  uint32x4 g0;
  g0.x = 1u;                                              // count=1, user mode
  g0.y = lds_addr;                                        // lds_addr [63:32]
  g0.z = (unsigned)(ga & 0xFFFFFFFFull);                  // global_addr[31:0]
  g0.w = (unsigned)((ga >> 32) & 0x01FFFFFFull) | (2u << 30); // addr[56:32], type=2
  int32x8 g1 = { 0x00010000,            // data_size=1 (2 bytes)
                 (int)(256u << 16),     // tensor_dim0 = 256 (bits 79:48)
                 (int)(4096u << 16),    // tensor_dim1 = 4096 (bits 111:80)
                 (int)(256u << 16),     // tile_dim0 = 256 (bits 127:112)
                 32,                    // tile_dim1 = 32 (bits 143:128)
                 256,                   // tensor_dim0_stride = 256 (bits 207:160)
                 0, 0 };
  int32x4 z4 = {0, 0, 0, 0};
#if __clang_major__ >= 23
  int32x8 z8 = {0, 0, 0, 0, 0, 0, 0, 0};
  __builtin_amdgcn_tensor_load_to_lds(g0, g1, z4, z4, z8, 0);
#else
  __builtin_amdgcn_tensor_load_to_lds(g0, g1, z4, z4, 0);
#endif
}
#endif

// ---------------------------------------------------------------------------
// Kernel 0a: transpose+convert x [B,C,N] f32 -> xt [B,N,C] bf16 (LDS tiled)
// ---------------------------------------------------------------------------
__global__ __launch_bounds__(256)
void xpose_bf16_kernel(const float* __restrict__ x, unsigned short* __restrict__ xt) {
  __shared__ unsigned short t[64][65];
  const int b = blockIdx.z, n0 = blockIdx.x * 64, c0 = blockIdx.y * 64;
  const float* xb = x + (size_t)b * C_ * N_;
  const int j = threadIdx.x & 63, i0 = threadIdx.x >> 6;
  for (int i = i0; i < 64; i += 4)                       // coalesced along n
    t[i][j] = f32_to_bf16(xb[(size_t)(c0 + i) * N_ + n0 + j]);
  __syncthreads();
  unsigned short* xtb = xt + (size_t)b * N_ * C_;
  for (int i = i0; i < 64; i += 4)                       // coalesced along c
    xtb[(size_t)(n0 + i) * C_ + c0 + j] = t[j][i];
}

// ---------------------------------------------------------------------------
// Kernel 0b: one-time f32 -> bf16 conversion of both weight matrices
// ---------------------------------------------------------------------------
__global__ __launch_bounds__(256)
void convert_w_kernel(const float* __restrict__ w_qkv, const float* __restrict__ w_out,
                      unsigned short* __restrict__ wq, unsigned short* __restrict__ wo) {
  int idx = blockIdx.x * 256 + threadIdx.x;
  if (idx < 3 * C_ * C_) wq[idx] = f32_to_bf16(w_qkv[idx]);
  if (idx < C_ * C_)     wo[idx] = f32_to_bf16(w_out[idx]);
}

// ---------------------------------------------------------------------------
// Kernel 1: QKV projection from bf16 operands. One wave -> 16(n) x 64(o) tile.
// Q pre-scaled by C^-0.5 = 1/16.
// ---------------------------------------------------------------------------
__global__ __launch_bounds__(32)
void qkv_proj_kernel(const unsigned short* __restrict__ xt,   // bf16 [B,N,C]
                     const unsigned short* __restrict__ wq,   // bf16 [3C,C]
                     const float* __restrict__ b_qkv,
                     unsigned short* __restrict__ q_ws, unsigned short* __restrict__ k_ws,
                     unsigned short* __restrict__ v_ws) {
  const int lane = threadIdx.x;
  const int nl = lane & 15, half = lane >> 4;
  const int n0 = blockIdx.x * 16;
  const int o0 = blockIdx.y * 64;
  const int b  = blockIdx.z;

  floatx8 acc[4] = {zero8(), zero8(), zero8(), zero8()};
  const unsigned int* arow =
      (const unsigned int*)(xt + ((size_t)b * N_ + n0 + nl) * C_);

  for (int k0 = 0; k0 < C_; k0 += 32) {
    FragBF a;
#pragma unroll
    for (int i = 0; i < 8; ++i)
      a.w[i] = arow[(k0 + a_kbase(i, half)) >> 1];
#pragma unroll
    for (int j = 0; j < 4; ++j) {
      FragBF bm;
      const unsigned int* wrow =
          (const unsigned int*)(wq + (size_t)(o0 + 16 * j + nl) * C_);
#pragma unroll
      for (int i = 0; i < 8; ++i)
        bm.w[i] = wrow[(k0 + b_kbase(i, half)) >> 1];
      acc[j] = __builtin_amdgcn_wmma_f32_16x16x32_bf16(false, a.v, false, bm.v,
                                                       (short)0, acc[j], false, false);
    }
  }

  const float qscale = 0.0625f; // C^-0.5, folded into Q
#pragma unroll
  for (int j = 0; j < 4; ++j) {
    int oc = o0 + 16 * j + nl;
    float bias = b_qkv[oc];
#pragma unroll
    for (int r = 0; r < 8; ++r) {
      int nrow = n0 + r + 8 * half;            // C/D layout: row = r + 8*half
      float val = acc[j][r] + bias;
      size_t rowbase = ((size_t)b * N_ + nrow) * C_;
      if (oc < C_)          q_ws[rowbase + oc]            = f32_to_bf16(val * qscale);
      else if (oc < 2 * C_) k_ws[rowbase + (oc - C_)]     = f32_to_bf16(val);
      else                  v_ws[rowbase + (oc - 2 * C_)] = f32_to_bf16(val);
    }
  }
}

// ---------------------------------------------------------------------------
// Kernel 2: fused flash attention, TDM-pipelined. 2 waves/WG, each owns 16
// query rows with a 16x256 f32 accumulator. K/V tiles (32x256 bf16) are
// double-buffered in LDS; the Tensor Data Mover fetches tile j+1 (wave0->K,
// wave1->V) while both waves run the WMMAs + online softmax of tile j.
// ---------------------------------------------------------------------------
__global__ __launch_bounds__(64)
void flash_attn_kernel(const unsigned short* __restrict__ q_ws,
                       const unsigned short* __restrict__ k_ws,
                       const unsigned short* __restrict__ v_ws,
                       unsigned short* __restrict__ attn_ws) {
  __shared__ unsigned short Qs[2][16 * C_];     // 16 KB  per-wave Q tile
  __shared__ unsigned short Ks[2][32 * C_];     // 32 KB  K tiles (double buffer)
  __shared__ unsigned short Vs[2][32 * C_];     // 32 KB  V tiles (double buffer)
  __shared__ unsigned short Ps[2][16 * 32];     //  2 KB  per-wave softmax probs

  const int tid  = threadIdx.x;
  const int wave = tid >> 5;
  const int lane = tid & 31;
  const int nl = lane & 15, half = lane >> 4;
  const int b  = blockIdx.z;
  const int n0 = (blockIdx.x * 2 + wave) * 16;
  const size_t bbase = (size_t)b * N_ * C_;

#ifdef USE_TDM
  const unsigned ks_lds[2] = {(unsigned)(size_t)(void*)Ks[0],
                              (unsigned)(size_t)(void*)Ks[1]};
  const unsigned vs_lds[2] = {(unsigned)(size_t)(void*)Vs[0],
                              (unsigned)(size_t)(void*)Vs[1]};
  // Prologue: wave0 fetches K(0); wave1 fetches the 32x256 Q tile and V(0).
  if (wave == 0) {
    tdm_load_tile32x256(ks_lds[0], k_ws + bbase);
  } else {
    tdm_load_tile32x256((unsigned)(size_t)(void*)Qs,
                        q_ws + bbase + (size_t)(blockIdx.x * 32) * C_);
    tdm_load_tile32x256(vs_lds[0], v_ws + bbase);
  }
#else
  {
    const unsigned int* qsrc =
        (const unsigned int*)(q_ws + bbase + (size_t)(blockIdx.x * 32) * C_);
    unsigned int* qdst = (unsigned int*)Qs;
    for (int p = tid; p < 32 * (C_ / 2); p += 64) qdst[p] = qsrc[p];
  }
#endif

  floatx8 o_acc[16];
#pragma unroll
  for (int i = 0; i < 16; ++i) o_acc[i] = zero8();
  float mrow[8], lrow[8];
#pragma unroll
  for (int r = 0; r < 8; ++r) { mrow[r] = -3.0e38f; lrow[r] = 0.f; }

  const unsigned int* Qs_u = (const unsigned int*)(Qs[wave]);

  for (int j0 = 0; j0 < N_; j0 += 32) {
    const int buf = (j0 >> 5) & 1;
#ifdef USE_TDM
    __builtin_amdgcn_s_wait_tensorcnt(0);   // tile j landed (issuing wave)
    __syncthreads();                        // publish LDS; retire readers of buf^1
    const int jn = j0 + 32;
    if (jn < N_) {                          // launch DMA of tile j+1 under compute
      if (wave == 0)
        tdm_load_tile32x256(ks_lds[buf ^ 1], k_ws + bbase + (size_t)jn * C_);
      else
        tdm_load_tile32x256(vs_lds[buf ^ 1], v_ws + bbase + (size_t)jn * C_);
    }
#else
    __syncthreads();
    {
      const unsigned int* ksrc = (const unsigned int*)(k_ws + bbase + (size_t)j0 * C_);
      const unsigned int* vsrc = (const unsigned int*)(v_ws + bbase + (size_t)j0 * C_);
      unsigned int* kdst = (unsigned int*)Ks[buf];
      unsigned int* vdst = (unsigned int*)Vs[buf];
      for (int p = tid; p < 32 * (C_ / 2); p += 64) kdst[p] = ksrc[p];
      for (int p = tid; p < 32 * (C_ / 2); p += 64) vdst[p] = vsrc[p];
    }
    __syncthreads();
#endif
    const unsigned int* Ks_u = (const unsigned int*)Ks[buf];
    const unsigned short* Vsb = Vs[buf];

    // S = Q * K^T  (16 queries x 32 keys), K-loop over C
    floatx8 s0 = zero8(), s1 = zero8();
#pragma unroll
    for (int kk = 0; kk < C_; kk += 32) {
      FragBF a, k0f, k1f;
#pragma unroll
      for (int i = 0; i < 8; ++i)
        a.w[i] = Qs_u[nl * (C_ / 2) + ((kk + a_kbase(i, half)) >> 1)];
#pragma unroll
      for (int i = 0; i < 8; ++i)
        k0f.w[i] = Ks_u[nl * (C_ / 2) + ((kk + b_kbase(i, half)) >> 1)];
      s0 = __builtin_amdgcn_wmma_f32_16x16x32_bf16(false, a.v, false, k0f.v,
                                                   (short)0, s0, false, false);
#pragma unroll
      for (int i = 0; i < 8; ++i)
        k1f.w[i] = Ks_u[(nl + 16) * (C_ / 2) + ((kk + b_kbase(i, half)) >> 1)];
      s1 = __builtin_amdgcn_wmma_f32_16x16x32_bf16(false, a.v, false, k1f.v,
                                                   (short)0, s1, false, false);
    }

    // Online softmax. Accumulator layout: row m = r + 8*half, col = nl.
    // Row-wise reductions = xor-shuffles within the 16-lane half-group.
#pragma unroll
    for (int r = 0; r < 8; ++r) {
      float v0 = s0[r], v1 = s1[r];
      float mx = fmaxf(v0, v1);
#pragma unroll
      for (int off = 8; off >= 1; off >>= 1)
        mx = fmaxf(mx, __shfl_xor(mx, off, 32));
      float mnew  = fmaxf(mrow[r], mx);
      float alpha = __expf(mrow[r] - mnew);
      float p0 = __expf(v0 - mnew);
      float p1 = __expf(v1 - mnew);
      float rs = p0 + p1;
#pragma unroll
      for (int off = 8; off >= 1; off >>= 1)
        rs += __shfl_xor(rs, off, 32);
      lrow[r] = lrow[r] * alpha + rs;
      mrow[r] = mnew;
#pragma unroll
      for (int cg = 0; cg < 16; ++cg) o_acc[cg][r] *= alpha;
      int m = r + 8 * half;
      Ps[wave][m * 32 + nl]      = f32_to_bf16(p0);
      Ps[wave][m * 32 + nl + 16] = f32_to_bf16(p1);
    }

    // P (accumulator layout in LDS) re-read as 16x32 A-fragment; O += P * V
    FragBF pfrag;
    const unsigned int* Ps_uw = (const unsigned int*)(Ps[wave]);
#pragma unroll
    for (int i = 0; i < 8; ++i)
      pfrag.w[i] = Ps_uw[nl * 16 + (a_kbase(i, half) >> 1)];
#pragma unroll
    for (int cg = 0; cg < 16; ++cg) {
      FragBF vf;
#pragma unroll
      for (int i = 0; i < 8; ++i) {
        int j = b_kbase(i, half);
        vf.u[2 * i]     = Vsb[j * C_ + cg * 16 + nl];
        vf.u[2 * i + 1] = Vsb[(j + 1) * C_ + cg * 16 + nl];
      }
      o_acc[cg] = __builtin_amdgcn_wmma_f32_16x16x32_bf16(false, pfrag.v, false, vf.v,
                                                          (short)0, o_acc[cg], false, false);
    }
  }

  // Normalize and emit attn[b,n,c] in bf16
#pragma unroll
  for (int r = 0; r < 8; ++r) {
    float inv = 1.0f / lrow[r];
    int nrow = n0 + r + 8 * half;
    size_t rowbase = bbase + (size_t)nrow * C_;
#pragma unroll
    for (int cg = 0; cg < 16; ++cg)
      attn_ws[rowbase + cg * 16 + nl] = f32_to_bf16(o_acc[cg][r] * inv);
  }
}

// ---------------------------------------------------------------------------
// Kernel 3: out projection + bias + residual, writing [B,C,H,W] layout.
// ---------------------------------------------------------------------------
__global__ __launch_bounds__(32)
void out_proj_kernel(const unsigned short* __restrict__ attn_ws,
                     const unsigned short* __restrict__ wo,    // bf16 [C,C]
                     const float* __restrict__ b_out,
                     const float* __restrict__ x, float* __restrict__ out) {
  const int lane = threadIdx.x;
  const int nl = lane & 15, half = lane >> 4;
  const int n0 = blockIdx.x * 16;
  const int o0 = blockIdx.y * 64;
  const int b  = blockIdx.z;
  const size_t bbase = (size_t)b * N_ * C_;

  floatx8 acc[4] = {zero8(), zero8(), zero8(), zero8()};
  const unsigned int* arow =
      (const unsigned int*)(attn_ws + bbase + (size_t)(n0 + nl) * C_);

  for (int k0 = 0; k0 < C_; k0 += 32) {
    FragBF a;
#pragma unroll
    for (int i = 0; i < 8; ++i)
      a.w[i] = arow[(k0 + a_kbase(i, half)) >> 1];
#pragma unroll
    for (int j = 0; j < 4; ++j) {
      FragBF bm;
      const unsigned int* wrow =
          (const unsigned int*)(wo + (size_t)(o0 + 16 * j + nl) * C_);
#pragma unroll
      for (int i = 0; i < 8; ++i)
        bm.w[i] = wrow[(k0 + b_kbase(i, half)) >> 1];
      acc[j] = __builtin_amdgcn_wmma_f32_16x16x32_bf16(false, a.v, false, bm.v,
                                                       (short)0, acc[j], false, false);
    }
  }

#pragma unroll
  for (int j = 0; j < 4; ++j) {
    int oc = o0 + 16 * j + nl;
    float bias = b_out[oc];
#pragma unroll
    for (int r = 0; r < 8; ++r) {
      int nrow = n0 + r + 8 * half;
      size_t idx = bbase + (size_t)oc * N_ + nrow;
      out[idx] = acc[j][r] + bias + x[idx];
    }
  }
}

// ---------------------------------------------------------------------------
extern "C" void kernel_launch(void* const* d_in, const int* in_sizes, int n_in,
                              void* d_out, int out_size, void* d_ws, size_t ws_size,
                              hipStream_t stream) {
  const float* x     = (const float*)d_in[0];
  const float* w_qkv = (const float*)d_in[1];
  const float* b_qkv = (const float*)d_in[2];
  const float* w_out = (const float*)d_in[3];
  const float* b_out = (const float*)d_in[4];
  float* out = (float*)d_out;

  const size_t per = (size_t)B_ * N_ * C_;                 // elements per tensor
  unsigned short* q_ws    = (unsigned short*)d_ws;         // bf16 [B,N,C], pre-scaled
  unsigned short* k_ws    = q_ws + per;
  unsigned short* v_ws    = k_ws + per;
  unsigned short* attn_ws = v_ws + per;
  unsigned short* xt_ws   = attn_ws + per;                 // bf16 [B,N,C]
  unsigned short* wq_ws   = xt_ws + per;                   // bf16 [3C,C]
  unsigned short* wo_ws   = wq_ws + (size_t)3 * C_ * C_;   // bf16 [C,C]
  // total workspace: ~42.5 MB

  dim3 g0a(N_ / 64, C_ / 64, B_);
  xpose_bf16_kernel<<<g0a, dim3(256), 0, stream>>>(x, xt_ws);
  convert_w_kernel<<<dim3((3 * C_ * C_ + 255) / 256), dim3(256), 0, stream>>>(
      w_qkv, w_out, wq_ws, wo_ws);

  dim3 g1(N_ / 16, (3 * C_) / 64, B_);
  qkv_proj_kernel<<<g1, dim3(32), 0, stream>>>(xt_ws, wq_ws, b_qkv, q_ws, k_ws, v_ws);

  dim3 g2(N_ / 32, 1, B_);
  flash_attn_kernel<<<g2, dim3(64), 0, stream>>>(q_ws, k_ws, v_ws, attn_ws);

  dim3 g3(N_ / 16, C_ / 64, B_);
  out_proj_kernel<<<g3, dim3(32), 0, stream>>>(attn_ws, wo_ws, b_out, x, out);
}
